// MaskGenerator_12000138625055
// MI455X (gfx1250) — matrix operand order repaired and checked by reading, not compiled, not run on hardware
//
#include <hip/hip_runtime.h>
#include <hip/hip_bf16.h>
#include <math.h>

// ---------------------------------------------------------------------------
// Types for CDNA5 WMMA
// ---------------------------------------------------------------------------
typedef __attribute__((ext_vector_type(16))) _Float16 v16h;
typedef __attribute__((ext_vector_type(8)))  _Float16 v8h;
typedef __attribute__((ext_vector_type(4)))  _Float16 v4h;
typedef __attribute__((ext_vector_type(2)))  _Float16 v2h;
typedef __attribute__((ext_vector_type(8)))  float    v8f;
typedef __attribute__((ext_vector_type(4)))  int      v4i;

#define BATCH 64
#define MAXF  256
#define DIM   1024
#define NHEAD 16
#define DH    64
#define DFF   4096
#define SMAX  16
#define GAMMA 0.1f

// CDNA5 async global->LDS copy path (ASYNCcnt), with synchronous fallback.
#if defined(__has_builtin)
#if __has_builtin(__builtin_amdgcn_global_load_async_to_lds_b128) && \
    __has_builtin(__builtin_amdgcn_s_wait_asynccnt)
#define HAVE_ASYNC_COPY 1
#endif
#endif

#define AS_GLOBAL __attribute__((address_space(1)))
#define AS_SHARED __attribute__((address_space(3)))

// ---------------------------------------------------------------------------
// Generic f32 -> f16 conversion (weights, attention outputs)
// ---------------------------------------------------------------------------
__global__ __launch_bounds__(256) void f32_to_f16_k(const float* __restrict__ s,
                                                    _Float16* __restrict__ d, int n) {
    int i = blockIdx.x * 256 + threadIdx.x;
    if (i < n) d[i] = (_Float16)s[i];
}

// memory = video_features + pe  (stored f16 for the CA K/V GEMMs)
__global__ __launch_bounds__(256) void build_mem16_k(const float* __restrict__ vf,
                                                     const float* __restrict__ pe,
                                                     _Float16* __restrict__ m16) {
    int i = blockIdx.x * 256 + threadIdx.x;
    if (i < BATCH * MAXF * DIM) {
        int fd = i % (MAXF * DIM);
        m16[i] = (_Float16)(vf[i] + pe[fd]);
    }
}

// queries = tpe[:16] * text_mask
__global__ __launch_bounds__(256) void build_queries_k(const float* __restrict__ tpe,
                                                       const long long* __restrict__ ns,
                                                       float* __restrict__ x) {
    int i = blockIdx.x * 256 + threadIdx.x;
    if (i < BATCH * SMAX * DIM) {
        int b  = i / (SMAX * DIM);
        int sd = i % (SMAX * DIM);
        int s  = sd / DIM;
        x[i] = (s < (int)ns[b]) ? tpe[sd] : 0.f;
    }
}

__global__ void zero_loss_k(float* p) { if (threadIdx.x == 0) *p = 0.f; }

// ---------------------------------------------------------------------------
// LayerNorm: one 256-thread block per row of D=1024.
// ---------------------------------------------------------------------------
__global__ __launch_bounds__(256)
void layernorm_k(const float* __restrict__ X, const float* __restrict__ g,
                 const float* __restrict__ bta, float* __restrict__ Yf,
                 _Float16* __restrict__ Yh, int D) {
    const int row = blockIdx.x, tid = threadIdx.x;
    const float* x = X + (size_t)row * D;
    __shared__ float red[256];
    float s = 0.f;
    for (int d = tid; d < D; d += 256) s += x[d];
    red[tid] = s; __syncthreads();
    for (int o = 128; o > 0; o >>= 1) { if (tid < o) red[tid] += red[tid + o]; __syncthreads(); }
    float mean = red[0] / (float)D; __syncthreads();
    float v = 0.f;
    for (int d = tid; d < D; d += 256) { float t = x[d] - mean; v += t * t; }
    red[tid] = v; __syncthreads();
    for (int o = 128; o > 0; o >>= 1) { if (tid < o) red[tid] += red[tid + o]; __syncthreads(); }
    float rstd = rsqrtf(red[0] / (float)D + 1e-5f); __syncthreads();
    for (int d = tid; d < D; d += 256) {
        float y = (x[d] - mean) * rstd * g[d] + bta[d];
        if (Yf) Yf[(size_t)row * D + d] = y;
        if (Yh) Yh[(size_t)row * D + d] = (_Float16)y;
    }
}

// ---------------------------------------------------------------------------
// WMMA GEMM: C[M,N] = act(A16[M,K] @ B16[K,N] + bias) (+ optional residual C)
//
// Block = 256 threads = 8 waves -> 64(M) x 64(N) C tile.
// Wave  = 16(M) x 32(N): one A fragment + two B fragments + 2 accumulators
//         -> 2x v_wmma_f32_16x16x32_f16 per K-step (K-step = 32).
//
// LDS staging:
//   sA : row-major 64x32 f16, row stride 40 halfs (80 B, keeps 16-B align).
//        A-frag = 16 contiguous halfs along k (k = 16*hi + s) -> 2x ds_load_b128.
//        Filled via async global->LDS b128 copies when available.
//   sBT: B tile stored TRANSPOSED (64 n x 32 k, stride 40). B-frag = 16
//        contiguous halfs along k -> 2x ds_load_b128. Filled with packed
//        b32 pair-stores from coalesced global b64 loads.
// ---------------------------------------------------------------------------
template <int ACT, bool ADD, bool OUT16>
__global__ __launch_bounds__(256)
void gemm_ws(const _Float16* __restrict__ A, const _Float16* __restrict__ B,
             const float* __restrict__ bias, float* __restrict__ Cf,
             _Float16* __restrict__ Ch, int M, int N, int K) {
    __shared__ _Float16 sA[64][40];    // 64 rows x 32 k (+pad)
    __shared__ _Float16 sBT[64][40];   // 64 n   x 32 k (+pad), transposed B
    const int tid  = threadIdx.x;
    const int m0   = blockIdx.y * 64;
    const int n0   = blockIdx.x * 64;
    const int w    = tid >> 5;
    const int lane = tid & 31;
    const int hi   = (lane >> 4) & 1;
    const int l    = lane & 15;
    const int wm   = (w >> 1) * 16;     // 0,16,32,48
    const int wn   = (w & 1) * 32;      // 0,32
    v8f acc0 = {0.f, 0.f, 0.f, 0.f, 0.f, 0.f, 0.f, 0.f};
    v8f acc1 = {0.f, 0.f, 0.f, 0.f, 0.f, 0.f, 0.f, 0.f};

    // per-thread load coordinates
    const int ar = tid >> 2;            // A: row 0..63
    const int ac = (tid & 3) * 8;       // A: k  0,8,16,24 (16-B aligned)
    const int bkp = tid >> 4;           // B: k-pair 0..15
    const int bng = (tid & 15) * 4;     // B: n group of 4

    for (int k0 = 0; k0 < K; k0 += 32) {
        // ---- stage A tile (64x32) : one b128 per thread ----
#ifdef HAVE_ASYNC_COPY
        __builtin_amdgcn_global_load_async_to_lds_b128(
            (AS_GLOBAL v4i*)(AS_GLOBAL void*)const_cast<_Float16*>(
                &A[(size_t)(m0 + ar) * K + k0 + ac]),
            (AS_SHARED v4i*)(AS_SHARED void*)(&sA[ar][ac]), 0, 0);
#else
        *(v8h*)&sA[ar][ac] = *(const v8h*)&A[(size_t)(m0 + ar) * K + k0 + ac];
#endif
        // ---- stage B tile (32x64) transposed into sBT ----
        {
            const int kk = 2 * bkp;
            const v4h r0 = *(const v4h*)&B[(size_t)(k0 + kk)     * N + n0 + bng];
            const v4h r1 = *(const v4h*)&B[(size_t)(k0 + kk + 1) * N + n0 + bng];
#pragma unroll
            for (int j = 0; j < 4; ++j) {
                v2h pr; pr.x = r0[j]; pr.y = r1[j];
                *(v2h*)&sBT[bng + j][kk] = pr;     // 4-B aligned packed store
            }
        }
        if (k0 + 32 < K)
            __builtin_prefetch(&B[(size_t)(k0 + 32 + bkp) * N + n0 + bng], 0, 1);
#ifdef HAVE_ASYNC_COPY
        __builtin_amdgcn_s_wait_asynccnt(0);
#endif
        __syncthreads();

        // ---- fragments: all aligned b128 LDS loads ----
        const v8h alo = *(const v8h*)&sA[wm + l][hi ? 16 : 0];
        const v8h ahi = *(const v8h*)&sA[wm + l][(hi ? 16 : 0) + 8];
        const v16h af = __builtin_shufflevector(alo, ahi,
            0, 1, 2, 3, 4, 5, 6, 7, 8, 9, 10, 11, 12, 13, 14, 15);

        const int kofs = hi ? 16 : 0;
        const v8h b0lo = *(const v8h*)&sBT[wn + l][kofs];
        const v8h b0hi = *(const v8h*)&sBT[wn + l][kofs + 8];
        const v16h bf0 = __builtin_shufflevector(b0lo, b0hi,
            0, 1, 2, 3, 4, 5, 6, 7, 8, 9, 10, 11, 12, 13, 14, 15);
        const v8h b1lo = *(const v8h*)&sBT[wn + 16 + l][kofs];
        const v8h b1hi = *(const v8h*)&sBT[wn + 16 + l][kofs + 8];
        const v16h bf1 = __builtin_shufflevector(b1lo, b1hi,
            0, 1, 2, 3, 4, 5, 6, 7, 8, 9, 10, 11, 12, 13, 14, 15);

        acc0 = __builtin_amdgcn_wmma_f32_16x16x32_f16(false, af, false, bf0,
                                                      (short)0, acc0, false, false);
        acc1 = __builtin_amdgcn_wmma_f32_16x16x32_f16(false, af, false, bf1,
                                                      (short)0, acc1, false, false);
        __syncthreads();
    }

    // ---- epilogue (C layout: VGPR r -> (M = r + hi*8, N = lane&15)) ----
    const int nn0 = n0 + wn + l;
    const int nn1 = nn0 + 16;
    const float b0 = bias ? bias[nn0] : 0.f;
    const float b1 = bias ? bias[nn1] : 0.f;
#pragma unroll
    for (int r = 0; r < 8; ++r) {
        const int mm = m0 + wm + r + hi * 8;
        float va = acc0[r] + b0;
        float vb = acc1[r] + b1;
        if (ACT == 1) {
            va = 0.5f * va * (1.f + erff(va * 0.70710678118f));
            vb = 0.5f * vb * (1.f + erff(vb * 0.70710678118f));
        }
        const size_t i0 = (size_t)mm * N + nn0;
        const size_t i1 = (size_t)mm * N + nn1;
        if (OUT16)    { Ch[i0]  = (_Float16)va; Ch[i1]  = (_Float16)vb; }
        else if (ADD) { Cf[i0] += va;           Cf[i1] += vb; }
        else          { Cf[i0]  = va;           Cf[i1]  = vb; }
    }
}

// ---------------------------------------------------------------------------
// Self-attention core: per (b,h) block, 16 queries x 16 keys, dh=64.
// ---------------------------------------------------------------------------
__global__ __launch_bounds__(256)
void attn_sa_k(const float* __restrict__ Q, const float* __restrict__ Kk,
               const float* __restrict__ V, const long long* __restrict__ ns_in,
               float* __restrict__ O) {
    const int b = blockIdx.x / NHEAD, h = blockIdx.x % NHEAD, tid = threadIdx.x;
    __shared__ float q[16][64], k[16][64], v[16][64], s[16][17];
    const int n = (int)ns_in[b];
    for (int e = tid; e < 16 * 64; e += 256) {
        int r = e >> 6, d = e & 63;
        size_t idx = ((size_t)(b * SMAX + r)) * DIM + h * DH + d;
        q[r][d] = Q[idx]; k[r][d] = Kk[idx]; v[r][d] = V[idx];
    }
    __syncthreads();
    {
        int qq = tid >> 4, kk = tid & 15;
        float acc = 0.f;
#pragma unroll 8
        for (int d = 0; d < 64; ++d) acc += q[qq][d] * k[kk][d];
        s[qq][kk] = (kk < n) ? acc * 0.125f : -1e9f;
    }
    __syncthreads();
    if (tid < 16) {
        float m = -1e30f;
        for (int kk = 0; kk < 16; ++kk) m = fmaxf(m, s[tid][kk]);
        float sum = 0.f;
        for (int kk = 0; kk < 16; ++kk) { float e = expf(s[tid][kk] - m); s[tid][kk] = e; sum += e; }
        float inv = 1.f / sum;
        for (int kk = 0; kk < 16; ++kk) s[tid][kk] *= inv;
    }
    __syncthreads();
    for (int e = tid; e < 16 * 64; e += 256) {
        int r = e >> 6, d = e & 63;
        float acc = 0.f;
#pragma unroll
        for (int kk = 0; kk < 16; ++kk) acc += s[r][kk] * v[kk][d];
        O[((size_t)(b * SMAX + r)) * DIM + h * DH + d] = acc;
    }
}

// ---------------------------------------------------------------------------
// Cross-attention core: per (b,h) block, 16 queries x 256 keys, dh=64.
// ---------------------------------------------------------------------------
__global__ __launch_bounds__(256)
void attn_ca_k(const float* __restrict__ Q, const _Float16* __restrict__ K16,
               const _Float16* __restrict__ V16, const unsigned char* __restrict__ fmask,
               float* __restrict__ O) {
    const int b = blockIdx.x / NHEAD, h = blockIdx.x % NHEAD, tid = threadIdx.x;
    __shared__ float q[16][64];
    __shared__ float s[16][257];
    __shared__ float red[256];
    for (int e = tid; e < 16 * 64; e += 256) {
        int r = e >> 6, d = e & 63;
        q[r][d] = Q[((size_t)(b * SMAX + r)) * DIM + h * DH + d];
    }
    __syncthreads();
    {   // thread tid owns key tid; K row loaded with b128 vector loads
        float kreg[64];
        const _Float16* kp = &K16[((size_t)(b * MAXF + tid)) * DIM + h * DH];
#pragma unroll
        for (int g = 0; g < 8; ++g) {
            v8h kv = *(const v8h*)&kp[g * 8];
#pragma unroll
            for (int j = 0; j < 8; ++j) kreg[g * 8 + j] = (float)kv[j];
        }
        bool valid = fmask[b * MAXF + tid] != 0;
        for (int qq = 0; qq < 16; ++qq) {
            float acc = 0.f;
#pragma unroll 8
            for (int d = 0; d < 64; ++d) acc += q[qq][d] * kreg[d];
            s[qq][tid] = valid ? acc * 0.125f : -1e9f;
        }
    }
    __syncthreads();
    for (int qq = 0; qq < 16; ++qq) {   // softmax over 256 keys
        float v = s[qq][tid];
        red[tid] = v; __syncthreads();
        for (int o = 128; o > 0; o >>= 1) { if (tid < o) red[tid] = fmaxf(red[tid], red[tid + o]); __syncthreads(); }
        float m = red[0]; __syncthreads();
        float e = expf(v - m);
        red[tid] = e; __syncthreads();
        for (int o = 128; o > 0; o >>= 1) { if (tid < o) red[tid] += red[tid + o]; __syncthreads(); }
        float sum = red[0]; __syncthreads();
        s[qq][tid] = e / sum;
        __syncthreads();
    }
    {   // O: thread -> (row r, 4 consecutive d); V read as b64 vectors
        const int r  = tid >> 4;
        const int d0 = (tid & 15) * 4;
        float a0 = 0.f, a1 = 0.f, a2 = 0.f, a3 = 0.f;
        const _Float16* vp = &V16[((size_t)(b * MAXF)) * DIM + h * DH + d0];
        for (int kk = 0; kk < MAXF; ++kk) {
            const float ws = s[r][kk];
            const v4h vv = *(const v4h*)&vp[(size_t)kk * DIM];
            a0 += ws * (float)vv.x; a1 += ws * (float)vv.y;
            a2 += ws * (float)vv.z; a3 += ws * (float)vv.w;
        }
        float* op = &O[((size_t)(b * SMAX + r)) * DIM + h * DH + d0];
        op[0] = a0; op[1] = a1; op[2] = a2; op[3] = a3;
    }
}

// ---------------------------------------------------------------------------
// Per-batch finalization: heads, sort, Gaussian masks, loss, ragged outputs.
// d_out layout: [0,T) lefts | [T,2T) rights | [2T] loss | [2T+1, 2T+1+256T) masks
// ---------------------------------------------------------------------------
__global__ __launch_bounds__(256)
void final_k(const float* __restrict__ outF, const float* __restrict__ wc,
             const float* __restrict__ ww, const long long* __restrict__ ns_in,
             const unsigned char* __restrict__ fmask, float* __restrict__ d_out, int T) {
    const int b = blockIdx.x, tid = threadIdx.x;
    __shared__ float red[256];
    __shared__ float cr[16], wr[16], sc[16], sw[16], nrmv[16];
    __shared__ float loc[16][257];
    __shared__ int sh_n, sh_off, sh_vl;
    if (tid == 0) {
        int off = 0;
        for (int i = 0; i < b; ++i) off += (int)ns_in[i];
        sh_off = off; sh_n = (int)ns_in[b];
    }
    __syncthreads();
    const int n = sh_n;
    for (int s = 0; s < 16; ++s) {
        const float* row = outF + ((size_t)(b * SMAX + s)) * DIM;
        float pc = 0.f, pw = 0.f;
        for (int d = tid; d < DIM; d += 256) { float v = row[d]; pc += v * wc[d]; pw += v * ww[d]; }
        red[tid] = pc; __syncthreads();
        for (int o = 128; o > 0; o >>= 1) { if (tid < o) red[tid] += red[tid + o]; __syncthreads(); }
        if (tid == 0) cr[s] = 1.f / (1.f + expf(-red[0]));
        __syncthreads();
        red[tid] = pw; __syncthreads();
        for (int o = 128; o > 0; o >>= 1) { if (tid < o) red[tid] += red[tid + o]; __syncthreads(); }
        if (tid == 0) wr[s] = 1.f / (1.f + expf(-red[0]));
        __syncthreads();
    }
    {
        red[tid] = (float)fmask[b * MAXF + tid]; __syncthreads();
        for (int o = 128; o > 0; o >>= 1) { if (tid < o) red[tid] += red[tid + o]; __syncthreads(); }
        if (tid == 0) sh_vl = (int)(red[0] + 0.5f);
        __syncthreads();
    }
    if (tid == 0) {   // sort by c_sent (padded sents pushed to end with key 2.0)
        float key[16]; int idx[16];
        for (int s = 0; s < 16; ++s) { key[s] = (s < n) ? cr[s] : 2.0f; idx[s] = s; }
        for (int i = 0; i < 16; ++i) {
            int mj = i;
            for (int j = i + 1; j < 16; ++j) if (key[j] < key[mj]) mj = j;
            float tk = key[i]; key[i] = key[mj]; key[mj] = tk;
            int ti = idx[i]; idx[i] = idx[mj]; idx[mj] = ti;
        }
        for (int i = 0; i < 16; ++i) { sc[i] = cr[idx[i]]; sw[i] = wr[idx[i]]; }
    }
    __syncthreads();
    {
        const int f = tid;
        const float vl = (float)sh_vl;
        const float tpos = (float)f / (vl - 1.f);
        const bool fv = fmask[b * MAXF + f] != 0;
        for (int s = 0; s < 16; ++s) {
            float c = sc[s], w = sw[s];
            float d = tpos - c;
            loc[s][f] = fv ? expf(-(d * d) / (2.f * w * w)) : 0.f;
        }
    }
    __syncthreads();
    for (int s = 0; s < 16; ++s) {
        float v = loc[s][tid];
        red[tid] = v * v; __syncthreads();
        for (int o = 128; o > 0; o >>= 1) { if (tid < o) red[tid] += red[tid + o]; __syncthreads(); }
        if (tid == 0) nrmv[s] = fmaxf(sqrtf(red[0]), 1e-12f);
        __syncthreads();
    }
    {
        int s = tid >> 4, t = tid & 15;
        float contrib = 0.f;
        if (s != t && s < n && t < n) {
            float dot = 0.f;
            for (int ff = 0; ff < MAXF; ++ff) dot += loc[s][ff] * loc[t][ff];
            float sim = dot / (nrmv[s] * nrmv[t]);
            contrib = fmaxf(sim - GAMMA, 0.f);
        }
        red[tid] = contrib; __syncthreads();
        for (int o = 128; o > 0; o >>= 1) { if (tid < o) red[tid] += red[tid + o]; __syncthreads(); }
        if (tid == 0) {
            float denom = (float)(n * (n - 1));
            atomicAdd(&d_out[2 * T], red[0] / denom / (float)BATCH);
        }
        __syncthreads();
    }
    float* lefts  = d_out;
    float* rights = d_out + T;
    float* masks  = d_out + 2 * T + 1;
    if (tid < n) {
        float c = sc[tid], w = sw[tid];
        lefts[sh_off + tid]  = fminf(fmaxf(c - 0.5f * w, 0.f), 1.f);
        rights[sh_off + tid] = fminf(fmaxf(c + 0.5f * w, 0.f), 1.f);
    }
    for (int e = tid; e < n * MAXF; e += 256) {
        int i = e >> 8, ff = e & 255;
        masks[(size_t)(sh_off + i) * MAXF + ff] = loc[i][ff];
    }
}

// ---------------------------------------------------------------------------
// Host-side launcher
// ---------------------------------------------------------------------------
extern "C" void kernel_launch(void* const* d_in, const int* in_sizes, int n_in,
                              void* d_out, int out_size, void* d_ws, size_t ws_size,
                              hipStream_t stream) {
    const float*         video = (const float*)d_in[0];
    const long long*     ns    = (const long long*)d_in[1];
    const unsigned char* fmask = (const unsigned char*)d_in[2];
    const float* tpe = (const float*)d_in[4];
    const float* pe  = (const float*)d_in[5];
    const float *sa_wq=(const float*)d_in[6],  *sa_bq=(const float*)d_in[7];
    const float *sa_wk=(const float*)d_in[8],  *sa_bk=(const float*)d_in[9];
    const float *sa_wv=(const float*)d_in[10], *sa_bv=(const float*)d_in[11];
    const float *sa_wo=(const float*)d_in[12], *sa_bo=(const float*)d_in[13];
    const float *ca_wq=(const float*)d_in[14], *ca_bq=(const float*)d_in[15];
    const float *ca_wk=(const float*)d_in[16], *ca_bk=(const float*)d_in[17];
    const float *ca_wv=(const float*)d_in[18], *ca_bv=(const float*)d_in[19];
    const float *ca_wo=(const float*)d_in[20], *ca_bo=(const float*)d_in[21];
    const float *ln1_g=(const float*)d_in[22], *ln1_b=(const float*)d_in[23];
    const float *ln2_g=(const float*)d_in[24], *ln2_b=(const float*)d_in[25];
    const float *ln3_g=(const float*)d_in[26], *ln3_b=(const float*)d_in[27];
    const float *lnf_g=(const float*)d_in[28], *lnf_b=(const float*)d_in[29];
    const float *ff_w1=(const float*)d_in[30], *ff_b1=(const float*)d_in[31];
    const float *ff_w2=(const float*)d_in[32], *ff_b2=(const float*)d_in[33];
    const float *w_center=(const float*)d_in[34], *w_width=(const float*)d_in[35];

    float* out = (float*)d_out;
    const int T = (out_size - 1) / 258;          // out_size = 258*T + 1

    char* p = (char*)d_ws;
    auto alloc = [&](size_t bytes) -> char* {
        char* r = p; p += (bytes + 255) & ~(size_t)255; return r;
    };
    const size_t DD = (size_t)DIM * DIM;
    const size_t TM = (size_t)BATCH * SMAX;      // 1024 token rows
    const size_t MM = (size_t)BATCH * MAXF;      // 16384 memory rows
    _Float16* w_saq16 = (_Float16*)alloc(DD * 2);
    _Float16* w_sak16 = (_Float16*)alloc(DD * 2);
    _Float16* w_sav16 = (_Float16*)alloc(DD * 2);
    _Float16* w_sao16 = (_Float16*)alloc(DD * 2);
    _Float16* w_caq16 = (_Float16*)alloc(DD * 2);
    _Float16* w_cak16 = (_Float16*)alloc(DD * 2);
    _Float16* w_cav16 = (_Float16*)alloc(DD * 2);
    _Float16* w_cao16 = (_Float16*)alloc(DD * 2);
    _Float16* w_ff1_16 = (_Float16*)alloc((size_t)DIM * DFF * 2);
    _Float16* w_ff2_16 = (_Float16*)alloc((size_t)DFF * DIM * 2);
    _Float16* mem16 = (_Float16*)alloc(MM * DIM * 2);
    float*    x     = (float*)alloc(TM * DIM * 4);
    _Float16* h16   = (_Float16*)alloc(TM * DIM * 2);
    float*    bufQ  = (float*)alloc(TM * DIM * 4);
    float*    bufK  = (float*)alloc(TM * DIM * 4);
    float*    bufV  = (float*)alloc(TM * DIM * 4);
    float*    bufO  = (float*)alloc(TM * DIM * 4);
    _Float16* o16   = (_Float16*)alloc(TM * DIM * 2);
    _Float16* Kc16  = (_Float16*)alloc(MM * DIM * 2);
    _Float16* Vc16  = (_Float16*)alloc(MM * DIM * 2);
    _Float16* u16   = (_Float16*)alloc(TM * DFF * 2);
    float*    outF  = bufK;   // reuse (free after self-attention)

    const int BT = 256;
    auto cgrid = [](size_t n) { return dim3((unsigned)((n + 255) / 256)); };

    // ---- 1. weight conversions to f16 ----
    f32_to_f16_k<<<cgrid(DD), BT, 0, stream>>>(sa_wq, w_saq16, (int)DD);
    f32_to_f16_k<<<cgrid(DD), BT, 0, stream>>>(sa_wk, w_sak16, (int)DD);
    f32_to_f16_k<<<cgrid(DD), BT, 0, stream>>>(sa_wv, w_sav16, (int)DD);
    f32_to_f16_k<<<cgrid(DD), BT, 0, stream>>>(sa_wo, w_sao16, (int)DD);
    f32_to_f16_k<<<cgrid(DD), BT, 0, stream>>>(ca_wq, w_caq16, (int)DD);
    f32_to_f16_k<<<cgrid(DD), BT, 0, stream>>>(ca_wk, w_cak16, (int)DD);
    f32_to_f16_k<<<cgrid(DD), BT, 0, stream>>>(ca_wv, w_cav16, (int)DD);
    f32_to_f16_k<<<cgrid(DD), BT, 0, stream>>>(ca_wo, w_cao16, (int)DD);
    f32_to_f16_k<<<cgrid((size_t)DIM * DFF), BT, 0, stream>>>(ff_w1, w_ff1_16, DIM * DFF);
    f32_to_f16_k<<<cgrid((size_t)DFF * DIM), BT, 0, stream>>>(ff_w2, w_ff2_16, DFF * DIM);

    // ---- 2. memory & queries ----
    build_mem16_k<<<cgrid(MM * DIM), BT, 0, stream>>>(video, pe, mem16);
    build_queries_k<<<cgrid(TM * DIM), BT, 0, stream>>>(tpe, ns, x);

    // ---- 3. self-attention block ----
    layernorm_k<<<(unsigned)TM, BT, 0, stream>>>(x, ln1_g, ln1_b, nullptr, h16, DIM);
    gemm_ws<0,false,false><<<dim3(DIM/64, (unsigned)TM/64), BT, 0, stream>>>(h16, w_saq16, sa_bq, bufQ, nullptr, (int)TM, DIM, DIM);
    gemm_ws<0,false,false><<<dim3(DIM/64, (unsigned)TM/64), BT, 0, stream>>>(h16, w_sak16, sa_bk, bufK, nullptr, (int)TM, DIM, DIM);
    gemm_ws<0,false,false><<<dim3(DIM/64, (unsigned)TM/64), BT, 0, stream>>>(h16, w_sav16, sa_bv, bufV, nullptr, (int)TM, DIM, DIM);
    attn_sa_k<<<BATCH * NHEAD, BT, 0, stream>>>(bufQ, bufK, bufV, ns, bufO);
    f32_to_f16_k<<<cgrid(TM * DIM), BT, 0, stream>>>(bufO, o16, (int)(TM * DIM));
    gemm_ws<0,true,false><<<dim3(DIM/64, (unsigned)TM/64), BT, 0, stream>>>(o16, w_sao16, sa_bo, x, nullptr, (int)TM, DIM, DIM);

    // ---- 4. cross-attention block ----
    layernorm_k<<<(unsigned)TM, BT, 0, stream>>>(x, ln2_g, ln2_b, nullptr, h16, DIM);
    gemm_ws<0,false,false><<<dim3(DIM/64, (unsigned)TM/64), BT, 0, stream>>>(h16, w_caq16, ca_bq, bufQ, nullptr, (int)TM, DIM, DIM);
    gemm_ws<0,false,true><<<dim3(DIM/64, (unsigned)MM/64), BT, 0, stream>>>(mem16, w_cak16, ca_bk, nullptr, Kc16, (int)MM, DIM, DIM);
    gemm_ws<0,false,true><<<dim3(DIM/64, (unsigned)MM/64), BT, 0, stream>>>(mem16, w_cav16, ca_bv, nullptr, Vc16, (int)MM, DIM, DIM);
    attn_ca_k<<<BATCH * NHEAD, BT, 0, stream>>>(bufQ, Kc16, Vc16, fmask, bufO);
    f32_to_f16_k<<<cgrid(TM * DIM), BT, 0, stream>>>(bufO, o16, (int)(TM * DIM));
    gemm_ws<0,true,false><<<dim3(DIM/64, (unsigned)TM/64), BT, 0, stream>>>(o16, w_cao16, ca_bo, x, nullptr, (int)TM, DIM, DIM);

    // ---- 5. FFN block ----
    layernorm_k<<<(unsigned)TM, BT, 0, stream>>>(x, ln3_g, ln3_b, nullptr, h16, DIM);
    gemm_ws<1,false,true><<<dim3(DFF/64, (unsigned)TM/64), BT, 0, stream>>>(h16, w_ff1_16, ff_b1, nullptr, u16, (int)TM, DFF, DIM);
    gemm_ws<0,true,false><<<dim3(DIM/64, (unsigned)TM/64), BT, 0, stream>>>(u16, w_ff2_16, ff_b2, x, nullptr, (int)TM, DIM, DFF);

    // ---- 6. final LN + heads + masks + loss + ragged outputs ----
    layernorm_k<<<(unsigned)TM, BT, 0, stream>>>(x, lnf_g, lnf_b, outF, nullptr, DIM);
    zero_loss_k<<<1, 32, 0, stream>>>(&out[2 * T]);
    final_k<<<BATCH, BT, 0, stream>>>(outF, w_center, w_width, ns, fmask, out, T);
}
